// LapPyrNet_68049461838210
// MI455X (gfx1250) — matrix operand order repaired and checked by reading, not compiled, not run on hardware
//
#include <hip/hip_runtime.h>
#include <hip/hip_bf16.h>
#include <math.h>

// ---------------------------------------------------------------------------
// MI455X (gfx1250) implementation.
// All heavy math is V_WMMA_F32_16X16X32_F16 implicit-GEMM conv: M=16 couts,
// N=16 pixels, K = Cin*9 in chunks of 32. Wave32, 4 waves/block sharing one
// LDS-staged input patch. Conv kernel is templated on CIN so every LDS/global
// fragment address folds to an immediate offset (no per-chunk division).
// DCNv2 = sampled modulated im2col in LDS + the same WMMA contraction.
// ---------------------------------------------------------------------------

typedef _Float16 half_t;
typedef half_t v8h  __attribute__((ext_vector_type(8)));
typedef half_t v16h __attribute__((ext_vector_type(16)));
typedef float  v8f  __attribute__((ext_vector_type(8)));

#define NI    10   // b*n images
#define NFR   5    // frames per batch
#define NFEAT 64
#define DGRP  8
#define KT    9

__device__ __forceinline__ float actf(float v, int act) {
  if (act == 1) return v > 0.f ? v : 0.f;
  if (act == 2) return v > 0.f ? v : 0.01f * v;
  return v;
}

// --- pack f32 OIHW 3x3 weights into f16 [CoutPad][tap*Cin + cin] -----------
__global__ void prep_w16(const float* __restrict__ w, half_t* __restrict__ o,
                         int Cout, int Cin, int CoutPad) {
  int total = CoutPad * Cin * 9;
  int i = blockIdx.x * 256 + threadIdx.x;
  if (i >= total) return;
  int K  = Cin * 9;
  int co = i / K;
  int k  = i - co * K;
  int t  = k / Cin;
  int ci = k - t * Cin;
  float v = (co < Cout) ? w[((size_t)co * Cin + ci) * 9 + t] : 0.f;
  o[(size_t)co * K + k] = (half_t)v;
}

// --- WMMA implicit-GEMM 3x3 conv, stride1 pad1, fused bias/act/residual ----
// src is (optionally) a channel-concat of src0 (C0 ch) and src1 (C1 ch).
// src0center: src0's image index is the batch-center frame (for 'cent' feats).
template <int CIN>
__global__ __launch_bounds__(128)
void conv3x3_wmma(const float* __restrict__ src0, const float* __restrict__ src1,
                  int C0,
                  const half_t* __restrict__ w16, const float* __restrict__ bias,
                  const float* __restrict__ res, float* __restrict__ out,
                  int H, int W, int Cout, int act, int src0center)
{
  __shared__ __align__(16) half_t sX[3 * 18 * CIN];   // [row][col][cin] f16
  const int HW  = H * W;
  const int tilesPerRow = W >> 4;
  const int tile = blockIdx.x;
  const int y  = tile / tilesPerRow;
  const int x0 = (tile - y * tilesPerRow) << 4;
  const int img  = blockIdx.z;
  const int img0 = src0center ? ((img / NFR) * NFR + NFR / 2) : img;
  const int tid  = threadIdx.x;

  // stage 3x18 halo patch for all CIN channels, f32 -> f16, zero-padded.
  // CIN*54 is divisible by 128; indices maintained incrementally (no div/mod
  // in the loop; 128 = 2*54 + 20).
  {
    constexpr int ITER = (CIN * 54) / 128;
    int cin = tid / 54;
    int rc  = tid - cin * 54;
    int r   = (rc >= 36) ? 2 : (rc >= 18 ? 1 : 0);
    int c   = rc - r * 18;
#pragma unroll
    for (int j = 0; j < ITER; ++j) {
      int gy = y + r - 1, gx = x0 + c - 1;
      float v = 0.f;
      if (gy >= 0 && gy < H && gx >= 0 && gx < W) {
        if (cin < C0) v = src0[((size_t)img0 * C0 + cin) * HW + gy * W + gx];
        else          v = src1[((size_t)img * (CIN - C0) + (cin - C0)) * HW + gy * W + gx];
      }
      sX[(r * 18 + c) * CIN + cin] = (half_t)v;
      // advance by 128 items: +2 cin, +20 positions
      c += 20; cin += 2;
      if (c >= 18) { c -= 18; r += 1; }
      if (c >= 18) { c -= 18; r += 1; }
      if (r >= 3)  { r -= 3; cin += 1; }
    }
  }
  __syncthreads();

  const int wave = tid >> 5;
  const int lane = tid & 31;
  const int CoutPad  = (Cout + 15) & ~15;
  const int coutBase = blockIdx.y * 64 + wave * 16;
  if (coutBase >= CoutPad) return;

  const int  n   = lane & 15;
  const bool hi  = lane >= 16;
  constexpr int Ktot   = CIN * 9;
  constexpr int chunks = Ktot >> 5;
  constexpr int cpt    = CIN >> 5;       // K-chunks per tap
  const int hiA = hi ? 8  : 0;
  const int hiB = hi ? 8  : 0;
  const half_t* wrow  = w16 + (size_t)(coutBase + n) * Ktot;  // A row M=lane&15
  const half_t* bbase = sX + n * CIN + hiB;                   // B col N=lane&15

  v8f acc = {0.f,0.f,0.f,0.f,0.f,0.f,0.f,0.f};
#pragma unroll
  for (int ck = 0; ck < chunks; ++ck) {
    constexpr_helper:;
    const int kb = ck << 5;
    v8h a0 = *(const v8h*)(wrow + kb + hiA);
    v8h a1 = *(const v8h*)(wrow + kb + 16 + hiA);
    v16h A = __builtin_shufflevector(a0, a1, 0,1,2,3,4,5,6,7,8,9,10,11,12,13,14,15);
    const int t  = ck / cpt;               // compile-time under unroll
    const int cb = (ck - t * cpt) << 5;
    const int dy = t / 3, dx = t - dy * 3;
    const half_t* bp = bbase + (dy * 18 + dx) * CIN + cb;
    v8h b0 = *(const v8h*)bp;
    v8h b1 = *(const v8h*)(bp + 16);
    v16h B = __builtin_shufflevector(b0, b1, 0,1,2,3,4,5,6,7,8,9,10,11,12,13,14,15);
    acc = __builtin_amdgcn_wmma_f32_16x16x32_f16(false, A, false, B,
                                                 (short)0, acc, false, false);
  }

  const int hiC = hi ? 8 : 0;
  const size_t pix = (size_t)y * W + x0 + n;
#pragma unroll
  for (int r = 0; r < 8; ++r) {
    int co = coutBase + r + hiC;
    if (co < Cout) {
      float v = actf(acc[r] + bias[co], act);
      size_t o = ((size_t)img * Cout + co) * HW + pix;
      if (res) v += res[o];
      out[o] = v;
    }
  }
}

// --- tiny 3->64 stem conv (direct, VALU) -----------------------------------
__global__ void conv_c0a(const float* __restrict__ x, const float* __restrict__ w,
                         const float* __restrict__ b, float* __restrict__ out,
                         int H, int W)
{
  int HW = H * W;
  size_t total = (size_t)NI * NFEAT * HW;
  size_t i = (size_t)blockIdx.x * blockDim.x + threadIdx.x;
  if (i >= total) return;
  int p = (int)(i % HW);
  size_t t2 = i / HW;
  int co  = (int)(t2 % NFEAT);
  int img = (int)(t2 / NFEAT);
  int yy = p / W, xx = p - yy * W;
  float acc = b[co];
  for (int ci = 0; ci < 3; ++ci) {
    const float* src = x + ((size_t)img * 3 + ci) * HW;
#pragma unroll
    for (int t = 0; t < 9; ++t) {
      int gy = yy + t / 3 - 1, gx = xx + t % 3 - 1;
      if (gy < 0 || gy >= H || gx < 0 || gx >= W) continue;
      acc += w[(co * 3 + ci) * 9 + t] * src[gy * W + gx];
    }
  }
  out[i] = acc;
}

// --- MaxPool2d(3,2,1) + LeakyReLU(0.01) ------------------------------------
__global__ void maxpool_lrelu(const float* __restrict__ in, float* __restrict__ out,
                              int H, int W)
{
  int H2 = H >> 1, W2 = W >> 1;
  size_t HW = (size_t)H * W, HW2 = (size_t)H2 * W2;
  size_t total = (size_t)NI * NFEAT * HW2;
  size_t i = (size_t)blockIdx.x * blockDim.x + threadIdx.x;
  if (i >= total) return;
  int p = (int)(i % HW2);
  size_t ic = i / HW2;
  int oy = p / W2, ox = p - oy * W2;
  const float* src = in + ic * HW;
  float m = -INFINITY;
#pragma unroll
  for (int dy = -1; dy <= 1; ++dy) {
    int iy = 2 * oy + dy; if (iy < 0 || iy >= H) continue;
#pragma unroll
    for (int dx = -1; dx <= 1; ++dx) {
      int ix = 2 * ox + dx; if (ix < 0 || ix >= W) continue;
      m = fmaxf(m, src[iy * W + ix]);
    }
  }
  out[i] = m > 0.f ? m : 0.01f * m;
}

// --- bilinear x2 upsample (half-pixel, clamp) and scale by 2.0 -------------
__global__ void upsample2x_scale(const float* __restrict__ in, float* __restrict__ out,
                                 int H, int W)
{
  int H2 = H * 2, W2 = W * 2;
  size_t HW = (size_t)H * W, HW2 = (size_t)H2 * W2;
  size_t total = (size_t)NI * NFEAT * HW2;
  size_t i = (size_t)blockIdx.x * blockDim.x + threadIdx.x;
  if (i >= total) return;
  int p = (int)(i % HW2);
  size_t ic = i / HW2;
  int yy = p / W2, xx = p - yy * W2;
  float sy = (yy + 0.5f) * 0.5f - 0.5f;
  float sx = (xx + 0.5f) * 0.5f - 0.5f;
  float fy = floorf(sy), fx = floorf(sx);
  float ly = sy - fy, lx = sx - fx;
  int y0 = (int)fy, x0 = (int)fx;
  int y0c = y0 < 0 ? 0 : (y0 > H - 1 ? H - 1 : y0);
  int y1c = y0 + 1 < 0 ? 0 : (y0 + 1 > H - 1 ? H - 1 : y0 + 1);
  int x0c = x0 < 0 ? 0 : (x0 > W - 1 ? W - 1 : x0);
  int x1c = x0 + 1 < 0 ? 0 : (x0 + 1 > W - 1 ? W - 1 : x0 + 1);
  const float* s = in + ic * HW;
  float v = (1.f - ly) * ((1.f - lx) * s[y0c * W + x0c] + lx * s[y0c * W + x1c])
          +        ly  * ((1.f - lx) * s[y1c * W + x0c] + lx * s[y1c * W + x1c]);
  out[i] = 2.f * v;
}

// --- DCNv2: sample modulated im2col into LDS, then WMMA contraction --------
__global__ __launch_bounds__(128)
void dcn_wmma(const float* __restrict__ x, const float* __restrict__ om,
              const half_t* __restrict__ w16, const float* __restrict__ bias,
              float* __restrict__ out, int H, int W)
{
  __shared__ __align__(16) half_t sS[KT * 16 * NFEAT];   // [tap][pixel][cin] f16
  const int HW = H * W;
  const int tilesPerRow = W >> 4;
  const int tile = blockIdx.x;
  const int y  = tile / tilesPerRow;
  const int x0 = (tile - y * tilesPerRow) << 4;
  const int img = blockIdx.z;
  const int tid = threadIdx.x;
  const float* xb = x  + (size_t)img * NFEAT * HW;
  const float* ob = om + (size_t)img * (3 * DGRP * KT) * HW;

#pragma unroll
  for (int j = 0; j < 9; ++j) {            // 1152 (pix,group,tap) items / 128
    int i = j * 128 + tid;
    int n    = i & 15;
    int rest = i >> 4;
    int g = rest & 7;
    int t = rest >> 3;
    int pxl = y * W + x0 + n;
    int ch  = g * KT + t;
    float dy = ob[(size_t)ch * HW + pxl];
    float dx = ob[(size_t)(72 + ch) * HW + pxl];
    float mk = ob[(size_t)(144 + ch) * HW + pxl];
    mk = 1.f / (1.f + __expf(-mk));
    float py = dy + (float)(t / 3 - 1) + (float)y;
    float px = dx + (float)(t % 3 - 1) + (float)(x0 + n);
    float fy = floorf(py), fx = floorf(px);
    float ly = py - fy, lx = px - fx;
    int iy0 = (int)fy, ix0 = (int)fx;
    float w00 = (1.f - ly) * (1.f - lx), w01 = (1.f - ly) * lx;
    float w10 = ly * (1.f - lx),         w11 = ly * lx;
    bool vy0 = (iy0 >= 0 && iy0 < H), vy1 = (iy0 + 1 >= 0 && iy0 + 1 < H);
    bool vx0 = (ix0 >= 0 && ix0 < W), vx1 = (ix0 + 1 >= 0 && ix0 + 1 < W);
#pragma unroll
    for (int c = 0; c < 8; ++c) {
      int cin = g * 8 + c;
      const float* pc = xb + (size_t)cin * HW;
      float v = 0.f;
      if (vy0) {
        if (vx0) v += w00 * pc[iy0 * W + ix0];
        if (vx1) v += w01 * pc[iy0 * W + ix0 + 1];
      }
      if (vy1) {
        if (vx0) v += w10 * pc[(iy0 + 1) * W + ix0];
        if (vx1) v += w11 * pc[(iy0 + 1) * W + ix0 + 1];
      }
      sS[((size_t)t * 16 + n) * NFEAT + cin] = (half_t)(v * mk);
    }
  }
  __syncthreads();

  const int wave = tid >> 5;
  const int lane = tid & 31;
  const int coutBase = wave * 16;
  const int  n  = lane & 15;
  const bool hi = lane >= 16;
  const int hiA = hi ? 8 : 0, hiB = hi ? 8 : 0, hiC = hi ? 8 : 0;
  const half_t* wrow  = w16 + (size_t)(coutBase + n) * 576;
  const half_t* bbase = sS + n * NFEAT + hiB;

  v8f acc = {0.f,0.f,0.f,0.f,0.f,0.f,0.f,0.f};
#pragma unroll
  for (int ck = 0; ck < 18; ++ck) {
    const int kb = ck << 5;
    v8h a0 = *(const v8h*)(wrow + kb + hiA);
    v8h a1 = *(const v8h*)(wrow + kb + 16 + hiA);
    v16h A = __builtin_shufflevector(a0, a1, 0,1,2,3,4,5,6,7,8,9,10,11,12,13,14,15);
    const int t  = ck >> 1;
    const int cb = (ck & 1) << 5;
    const half_t* bp = bbase + t * 16 * NFEAT + cb;
    v8h b0 = *(const v8h*)bp;
    v8h b1 = *(const v8h*)(bp + 16);
    v16h B = __builtin_shufflevector(b0, b1, 0,1,2,3,4,5,6,7,8,9,10,11,12,13,14,15);
    acc = __builtin_amdgcn_wmma_f32_16x16x32_f16(false, A, false, B,
                                                 (short)0, acc, false, false);
  }

  const size_t pix = (size_t)y * W + x0 + n;
#pragma unroll
  for (int r = 0; r < 8; ++r) {
    int co = coutBase + r + hiC;
    out[((size_t)img * NFEAT + co) * HW + pix] = acc[r] + bias[co];
  }
}

// ---------------------------------------------------------------------------
extern "C" void kernel_launch(void* const* d_in, const int* in_sizes, int n_in,
                              void* d_out, int out_size, void* d_ws, size_t ws_size,
                              hipStream_t stream)
{
  const float* images = (const float*)d_in[0];
  auto Pw = [&](int i) { return (const float*)d_in[i]; };

  const float *c0a_w=Pw(1),  *c0a_b=Pw(2);
  const float *c0rb_c1w=Pw(3),  *c0rb_c1b=Pw(4),  *c0rb_c2w=Pw(5),  *c0rb_c2b=Pw(6);
  const float *c0b_w=Pw(7),  *c0b_b=Pw(8);
  const float *drb1_c1w=Pw(9),  *drb1_c1b=Pw(10), *drb1_c2w=Pw(11), *drb1_c2b=Pw(12);
  const float *drb2_c1w=Pw(13), *drb2_c1b=Pw(14), *drb2_c2w=Pw(15), *drb2_c2b=Pw(16);
  const float *down_w=Pw(17), *down_b=Pw(18);
  const float *off_in_w=Pw(19), *off_in_b=Pw(20);
  const float *offrb_c1w=Pw(21),*offrb_c1b=Pw(22),*offrb_c2w=Pw(23),*offrb_c2b=Pw(24);
  const float *off_out_w=Pw(25),*off_out_b=Pw(26);
  const float *mrg_in_w=Pw(27), *mrg_in_b=Pw(28);
  const float *mrgrb_c1w=Pw(29),*mrgrb_c1b=Pw(30),*mrgrb_c2w=Pw(31),*mrgrb_c2b=Pw(32);
  const float *mrg_out_w=Pw(33),*mrg_out_b=Pw(34);
  const float *om_w=Pw(35), *om_b=Pw(36);
  const float *dcn_w=Pw(37), *dcn_b=Pw(38);

  char* base = (char*)d_ws;
  size_t cur = 0;
  auto carve = [&](size_t bytes) -> void* {
    void* p = base + cur;
    cur += (bytes + 255) & ~(size_t)255;
    return p;
  };

  const int H0=128, W0=128, H1=64, W1=64, H2=32, W2=32;
  const size_t S0 = (size_t)NI * 64 * H0 * W0;
  const size_t S1 = (size_t)NI * 64 * H1 * W1;
  const size_t S2 = (size_t)NI * 64 * H2 * W2;

  const size_t WB64  = (size_t)64  * 576  * 2;   // Cin=64  Cout=64 packed f16
  const size_t WB128 = (size_t)64  * 1152 * 2;   // Cin=128 Cout=64
  const size_t WBOM  = (size_t)224 * 576  * 2;   // Cin=64  Cout=216 (pad 224)

  half_t* w16_c0rb1 = (half_t*)carve(WB64);
  half_t* w16_c0rb2 = (half_t*)carve(WB64);
  half_t* w16_c0b   = (half_t*)carve(WB64);
  half_t* w16_drb1a = (half_t*)carve(WB64);
  half_t* w16_drb1b = (half_t*)carve(WB64);
  half_t* w16_drb2a = (half_t*)carve(WB64);
  half_t* w16_drb2b = (half_t*)carve(WB64);
  half_t* w16_down  = (half_t*)carve(WB64);
  half_t* w16_offin = (half_t*)carve(WB128);
  half_t* w16_offr1 = (half_t*)carve(WB64);
  half_t* w16_offr2 = (half_t*)carve(WB64);
  half_t* w16_offo  = (half_t*)carve(WB64);
  half_t* w16_mrgin = (half_t*)carve(WB128);
  half_t* w16_mrgr1 = (half_t*)carve(WB64);
  half_t* w16_mrgr2 = (half_t*)carve(WB64);
  half_t* w16_mrgo  = (half_t*)carve(WB64);
  half_t* w16_om    = (half_t*)carve(WBOM);
  half_t* w16_dcn   = (half_t*)carve(WB64);

  float* pyr0  = (float*)carve(S0 * 4);
  float* pyr1  = (float*)carve(S1 * 4);
  float* pyr2  = (float*)carve(S2 * 4);
  float* tmpA  = (float*)carve(S0 * 4);
  float* tmpB  = (float*)carve(S0 * 4);
  float* offL2 = (float*)carve(S2 * 4);
  float* offL1 = (float*)carve(S1 * 4);
  float* offL0 = (float*)carve(S0 * 4);
  float* offT  = (float*)carve(S0 * 4);    // pre-merge 'off' branch
  float* upB   = (float*)carve(S0 * 4);
  float* omB   = (float*)carve((size_t)NI * 216 * H0 * W0 * 4);

  auto prep = [&](const float* w, half_t* o, int Cout, int Cin) {
    int CoutPad = (Cout + 15) & ~15;
    int total = CoutPad * Cin * 9;
    prep_w16<<<(total + 255) / 256, 256, 0, stream>>>(w, o, Cout, Cin, CoutPad);
  };
  prep(c0rb_c1w, w16_c0rb1, 64, 64);  prep(c0rb_c2w, w16_c0rb2, 64, 64);
  prep(c0b_w,    w16_c0b,   64, 64);
  prep(drb1_c1w, w16_drb1a, 64, 64);  prep(drb1_c2w, w16_drb1b, 64, 64);
  prep(drb2_c1w, w16_drb2a, 64, 64);  prep(drb2_c2w, w16_drb2b, 64, 64);
  prep(down_w,   w16_down,  64, 64);
  prep(off_in_w, w16_offin, 64, 128);
  prep(offrb_c1w,w16_offr1, 64, 64);  prep(offrb_c2w,w16_offr2, 64, 64);
  prep(off_out_w,w16_offo,  64, 64);
  prep(mrg_in_w, w16_mrgin, 64, 128);
  prep(mrgrb_c1w,w16_mrgr1, 64, 64);  prep(mrgrb_c2w,w16_mrgr2, 64, 64);
  prep(mrg_out_w,w16_mrgo,  64, 64);
  prep(om_w,     w16_om,   216, 64);
  prep(dcn_w,    w16_dcn,   64, 64);

  auto conv = [&](const float* s0, const float* s1, int C0, int C1,
                  const half_t* w, const float* b, const float* res, float* o,
                  int H, int W, int Cout, int act, int cent) {
    int CoutPad = (Cout + 15) & ~15;
    dim3 g((unsigned)(H * (W >> 4)), (unsigned)((CoutPad + 63) / 64), NI);
    if (C0 + C1 == 128)
      conv3x3_wmma<128><<<g, 128, 0, stream>>>(s0, s1, C0, w, b, res, o,
                                               H, W, Cout, act, cent);
    else
      conv3x3_wmma<64><<<g, 128, 0, stream>>>(s0, s1, C0, w, b, res, o,
                                              H, W, Cout, act, cent);
  };

  // ---- shared feature trunk ----
  {
    size_t total = (size_t)NI * 64 * H0 * W0;
    conv_c0a<<<(unsigned)((total + 255) / 256), 256, 0, stream>>>(
        images, c0a_w, c0a_b, tmpA, H0, W0);
  }
  conv(tmpA, nullptr, 64, 0, w16_c0rb1, c0rb_c1b, nullptr, tmpB, H0, W0, 64, 1, 0);
  conv(tmpB, nullptr, 64, 0, w16_c0rb2, c0rb_c2b, tmpA,    tmpA, H0, W0, 64, 0, 0);
  conv(tmpA, nullptr, 64, 0, w16_c0b,   c0b_b,    nullptr, pyr0, H0, W0, 64, 2, 0);

  auto down = [&](const float* src, float* dst, int H, int W) {
    conv(src,  nullptr, 64, 0, w16_drb1a, drb1_c1b, nullptr, tmpB, H, W, 64, 1, 0);
    conv(tmpB, nullptr, 64, 0, w16_drb1b, drb1_c2b, src,     tmpA, H, W, 64, 0, 0);
    conv(tmpA, nullptr, 64, 0, w16_drb2a, drb2_c1b, nullptr, tmpB, H, W, 64, 1, 0);
    conv(tmpB, nullptr, 64, 0, w16_drb2b, drb2_c2b, tmpA,    tmpA, H, W, 64, 0, 0);
    conv(tmpA, nullptr, 64, 0, w16_down,  down_b,   nullptr, tmpB, H, W, 64, 0, 0);
    size_t total = (size_t)NI * 64 * (H >> 1) * (W >> 1);
    maxpool_lrelu<<<(unsigned)((total + 255) / 256), 256, 0, stream>>>(tmpB, dst, H, W);
  };
  down(pyr0, pyr1, H0, W0);
  down(pyr1, pyr2, H1, W1);

  // offset nets (resblock applied twice with shared weights, as in reference)
  auto offnet = [&](const float* s0, const float* s1, int cent, float* dst,
                    const half_t* win, const float* bin,
                    const half_t* wr1, const float* br1,
                    const half_t* wr2, const float* br2,
                    const half_t* wout, const float* bout, int H, int W) {
    conv(s0, s1, 64, 64, win, bin, nullptr, tmpA, H, W, 64, 2, cent);
    for (int it = 0; it < 2; ++it) {
      conv(tmpA, nullptr, 64, 0, wr1, br1, nullptr, tmpB, H, W, 64, 1, 0);
      conv(tmpB, nullptr, 64, 0, wr2, br2, tmpA,    tmpA, H, W, 64, 0, 0);
    }
    conv(tmpA, nullptr, 64, 0, wout, bout, nullptr, dst, H, W, 64, 2, 0);
  };

  float* lap0 = (float*)d_out;
  float* lap1 = lap0 + S0;
  float* lap2 = lap1 + S1;

  auto dcn = [&](const float* x, const float* feat, float* o, int H, int W) {
    conv(feat, nullptr, 64, 0, w16_om, om_b, nullptr, omB, H, W, 216, 0, 0);
    dim3 g((unsigned)(H * (W >> 4)), 1, NI);
    dcn_wmma<<<g, 128, 0, stream>>>(x, omB, w16_dcn, dcn_b, o, H, W);
  };

  // ---- level 2 (coarsest) ----
  offnet(pyr2, pyr2, 1, offL2, w16_offin, off_in_b, w16_offr1, offrb_c1b,
         w16_offr2, offrb_c2b, w16_offo, off_out_b, H2, W2);
  dcn(pyr2, offL2, lap2, H2, W2);

  // ---- level 1 ----
  offnet(pyr1, pyr1, 1, offT, w16_offin, off_in_b, w16_offr1, offrb_c1b,
         w16_offr2, offrb_c2b, w16_offo, off_out_b, H1, W1);
  {
    size_t total = (size_t)NI * 64 * H1 * W1;
    upsample2x_scale<<<(unsigned)((total + 255) / 256), 256, 0, stream>>>(offL2, upB, H2, W2);
  }
  offnet(offT, upB, 0, offL1, w16_mrgin, mrg_in_b, w16_mrgr1, mrgrb_c1b,
         w16_mrgr2, mrgrb_c2b, w16_mrgo, mrg_out_b, H1, W1);
  dcn(pyr1, offL1, lap1, H1, W1);

  // ---- level 0 ----
  offnet(pyr0, pyr0, 1, offT, w16_offin, off_in_b, w16_offr1, offrb_c1b,
         w16_offr2, offrb_c2b, w16_offo, off_out_b, H0, W0);
  {
    size_t total = (size_t)NI * 64 * H0 * W0;
    upsample2x_scale<<<(unsigned)((total + 255) / 256), 256, 0, stream>>>(offL1, upB, H1, W1);
  }
  offnet(offT, upB, 0, offL0, w16_mrgin, mrg_in_b, w16_mrgr1, mrgrb_c1b,
         w16_mrgr2, mrgrb_c2b, w16_mrgo, mrg_out_b, H0, W0);
  dcn(pyr0, offL0, lap0, H0, W0);
}
